// TopKDecorator_67843303408227
// MI455X (gfx1250) — matrix operand order repaired and checked by reading, not compiled, not run on hardware
//
#include <hip/hip_runtime.h>
#include <hip/hip_fp16.h>

// ---------------- problem constants (match reference) ----------------
#define TOPK      21
#define N_ITEMS   100000
#define EMBED_DIM 128
#define BATCH     4096
#define HIST_LEN  50

// ---------------- tiling ----------------
#define NWAVES  8                 // 256 threads / wave32
#define BM      32                // batch rows per workgroup (2 WMMA M-tiles)
#define N_SPLIT 2                 // item-table splits (keeps grid at 256 WGs)
#define ITEMS_PER_SPLIT ((N_ITEMS + N_SPLIT - 1) / N_SPLIT)
// each wave covers 16 items/chunk for BOTH row groups -> 8 WMMAs : 8 loads

typedef __attribute__((ext_vector_type(16))) _Float16 v16h;
typedef __attribute__((ext_vector_type(8)))  float    v8f;

// ---------------------------------------------------------------------
// Kernel 1: convert the item table f32 -> f16 (25.6 MB, lives in L2).
// ---------------------------------------------------------------------
__global__ void cvt_f16_kernel(const float* __restrict__ src,
                               _Float16* __restrict__ dst, int n) {
  int i = blockIdx.x * blockDim.x + threadIdx.x;
  int stride = gridDim.x * blockDim.x;
  for (; i < n; i += stride) dst[i] = (_Float16)src[i];
}

// ---------------------------------------------------------------------
// Kernel 2: masked mean-pool session encoder -> query [B, D], stored f16.
// ---------------------------------------------------------------------
__global__ void query_kernel(const int* __restrict__ item_seq,
                             const int* __restrict__ item_seq_len,
                             const float* __restrict__ emb,
                             _Float16* __restrict__ qh) {
  const int row = blockIdx.x;
  const int d   = threadIdx.x;
  const int len = item_seq_len[row];
  float s = 0.f;
  for (int t = 0; t < len; ++t) {
    const int it = item_seq[row * HIST_LEN + t];
    s += emb[(size_t)it * EMBED_DIM + d];
  }
  const float denom = (float)max(len, 1);
  qh[(size_t)row * EMBED_DIM + d] = (_Float16)(s / denom);
}

// ---------------------------------------------------------------------
// Kernel 3: fused  scores = Q @ E^T  (WMMA f16->f32)  +  running top-21.
//
// 8 waves share a 32-row tile; each wave holds TWO A fragments and feeds
// each B fragment to 2 WMMAs (8 WMMAs : 8 b128 loads), halving L2 B
// traffic vs BM=16. B is double-buffered; thresholds live in registers;
// one ballot per iteration gates the rare insert path. Each (row-tile,
// item-split) workgroup emits a sorted partial top-21 to workspace.
// ---------------------------------------------------------------------
__launch_bounds__(256, 1)
__global__ void score_topk_kernel(const _Float16* __restrict__ embh,
                                  const _Float16* __restrict__ qh,
                                  float* __restrict__ pv,
                                  int*   __restrict__ pi) {
  __shared__ float s_val[NWAVES][BM][TOPK];   // 21.5 KB
  __shared__ int   s_idx[NWAVES][BM][TOPK];   // 21.5 KB

  const int tid   = threadIdx.x;
  const int wave  = tid >> 5;
  const int lane  = tid & 31;
  const int khalf = lane >> 4;   // wave half (selects K group)
  const int nsub  = lane & 15;   // row/col within the 16-wide tile
  const int rowBase   = blockIdx.x * BM;
  const int split     = blockIdx.y;
  const int itemStart = split * ITEMS_PER_SPLIT;
  const int itemEnd   = min(N_ITEMS, itemStart + ITEMS_PER_SPLIT);

  // init per-wave top-k lists (ascending; [0] is the running threshold)
  for (int i = tid; i < NWAVES * BM * TOPK; i += 256) {
    ((float*)s_val)[i] = -INFINITY;
    ((int*)s_idx)[i]   = 0;
  }
  __syncthreads();

  // ---- A fragments for both row groups, loaded once.
  // A k order used here (contiguous, not the ISA table order): lane<16 holds M=lane, K in {0..15};
  // lane>=16 holds M=lane-16, K in {16..31}; pairs are contiguous.
  union AU { v16h v; unsigned int u[8]; };
  AU a0[4], a1[4];
  {
    const _Float16* q0 = qh + (size_t)(rowBase + nsub) * EMBED_DIM;
    const _Float16* q1 = q0 + 16 * EMBED_DIM;
    #pragma unroll
    for (int ks = 0; ks < 4; ++ks) {
      #pragma unroll
      for (int j = 0; j < 8; ++j) {
        const int koff = ks * 32 + 2 * j + khalf * 16;
        a0[ks].u[j] = *(const unsigned int*)(q0 + koff);
        a1[ks].u[j] = *(const unsigned int*)(q1 + koff);
      }
    }
  }

  const int itemsPerIter = NWAVES * 16;   // 128 items per workgroup iteration

  // B k order used here (same contiguous order as A): lane n = lane%16 is the item (N); lane<16 holds
  // K=0..15 of the k-step, lane>=16 holds K=16..31 -> contiguous 32B loads.
  auto bptr = [&](int base) -> const _Float16* {
    const int item  = base + wave * 16 + nsub;
    const int itemc = min(item, N_ITEMS - 1);
    return embh + (size_t)itemc * EMBED_DIM + khalf * 16;
  };

  // per-row threshold cache (monotonically rising; stale-low is safe)
  float thrReg[16];
  #pragma unroll
  for (int r = 0; r < 16; ++r) thrReg[r] = -INFINITY;

  // prime the B pipeline
  v16h bcur[4];
  {
    const _Float16* p = bptr(itemStart);
    #pragma unroll
    for (int ks = 0; ks < 4; ++ks) bcur[ks] = *(const v16h*)(p + ks * 32);
  }

  for (int base = itemStart; base < itemEnd; base += itemsPerIter) {
    const int nbase = base + itemsPerIter;
    // ---- issue next chunk's B loads first (overlap with WMMA + harvest)
    v16h bnxt[4];
    {
      const _Float16* p = bptr(nbase < itemEnd ? nbase : itemStart);
      __builtin_prefetch(p + (size_t)itemsPerIter * EMBED_DIM, 0, 3);
      #pragma unroll
      for (int ks = 0; ks < 4; ++ks) bnxt[ks] = *(const v16h*)(p + ks * 32);
    }

    // ---- 8 WMMAs on resident data; c0/c1 independent -> no RAW stalls
    v8f c0 = {0.f, 0.f, 0.f, 0.f, 0.f, 0.f, 0.f, 0.f};
    v8f c1 = {0.f, 0.f, 0.f, 0.f, 0.f, 0.f, 0.f, 0.f};
    #pragma unroll
    for (int ks = 0; ks < 4; ++ks) {
      c0 = __builtin_amdgcn_wmma_f32_16x16x32_f16(
               false, a0[ks].v, false, bcur[ks], (short)0, c0, false, false);
      c1 = __builtin_amdgcn_wmma_f32_16x16x32_f16(
               false, a1[ks].v, false, bcur[ks], (short)0, c1, false, false);
    }

    // ---- harvest. C/D layout: VGPR r holds row M = r + 8*khalf (+16 for c1),
    // column N = lane%16. Common case: 16 compares + 1 ballot, no LDS.
    const bool valid = (base + wave * 16 + nsub) < itemEnd;
    bool any = false;
    #pragma unroll
    for (int r = 0; r < 8; ++r) {
      any |= (c0[r] > thrReg[r]);
      any |= (c1[r] > thrReg[8 + r]);
    }

    if (__ballot(valid && any)) {
      #pragma unroll
      for (int g = 0; g < 2; ++g) {
        #pragma unroll
        for (int r = 0; r < 8; ++r) {
          const float cv = g ? c1[r] : c0[r];
          unsigned long long msk = __ballot(valid && (cv > thrReg[g * 8 + r]));
          while (msk) {                      // wave-uniform loop
            const int src = __ffsll(msk) - 1;
            msk &= msk - 1;
            const float v = __shfl(cv, src, 32);
            if (lane == 0) {                 // serial insert, rare
              const int srow  = g * 16 + r + ((src >> 4) << 3);
              const int sitem = base + wave * 16 + (src & 15);
              float* vals = s_val[wave][srow];
              int*   idxs = s_idx[wave][srow];
              if (v > vals[0]) {             // re-check vs true threshold
                int i = 0;
                #pragma clang loop unroll(disable)
                while (i + 1 < TOPK && vals[i + 1] < v) {
                  vals[i] = vals[i + 1];
                  idxs[i] = idxs[i + 1];
                  ++i;
                }
                vals[i] = v;
                idxs[i] = sitem;
              }
            }
          }
        }
      }
      // refresh register thresholds (uniform path; lists only got tighter)
      #pragma unroll
      for (int r = 0; r < 8; ++r) {
        thrReg[r]     = s_val[wave][r + 8 * khalf][0];
        thrReg[8 + r] = s_val[wave][16 + r + 8 * khalf][0];
      }
    }

    // ---- rotate the B double buffer
    #pragma unroll
    for (int ks = 0; ks < 4; ++ks) bcur[ks] = bnxt[ks];
  }
  __syncthreads();

  // ---- merge 8 per-wave ascending lists per row -> sorted partial top-21
  if (tid < BM) {
    const int m = tid;
    int cur[NWAVES];
    #pragma unroll
    for (int w = 0; w < NWAVES; ++w) cur[w] = TOPK - 1;
    const size_t outRow = ((size_t)split * BATCH + (rowBase + m)) * TOPK;
    #pragma clang loop unroll(disable)
    for (int t = 0; t < TOPK; ++t) {
      float best = -INFINITY;
      int bw = 0;
      #pragma unroll
      for (int w = 0; w < NWAVES; ++w) {
        if (cur[w] >= 0) {
          const float v = s_val[w][m][cur[w]];
          if (v > best) { best = v; bw = w; }
        }
      }
      pv[outRow + t] = best;
      pi[outRow + t] = s_idx[bw][m][cur[bw]];
      cur[bw]--;
    }
  }
}

// ---------------------------------------------------------------------
// Kernel 4: final 2-way merge of the per-split descending top-21 lists.
// Split 0 wins ties (lower item indices), matching jax.lax.top_k.
// ---------------------------------------------------------------------
__global__ void merge_kernel(const float* __restrict__ pv,
                             const int*   __restrict__ pi,
                             float* __restrict__ out) {
  const int row = blockIdx.x * blockDim.x + threadIdx.x;
  if (row >= BATCH) return;
  const float* v0 = pv + (size_t)row * TOPK;
  const float* v1 = pv + ((size_t)BATCH + row) * TOPK;
  const int*   i0 = pi + (size_t)row * TOPK;
  const int*   i1 = pi + ((size_t)BATCH + row) * TOPK;
  int p0 = 0, p1 = 0;
  #pragma clang loop unroll(disable)
  for (int t = 0; t < TOPK; ++t) {
    const float a = v0[p0], b = v1[p1];
    const bool takeA = (a >= b);
    out[(size_t)row * TOPK + t] = takeA ? a : b;
    out[(size_t)BATCH * TOPK + (size_t)row * TOPK + t] =
        (float)(takeA ? i0[p0] : i1[p1]);
    if (takeA) ++p0; else ++p1;
  }
}

// ---------------------------------------------------------------------
extern "C" void kernel_launch(void* const* d_in, const int* in_sizes, int n_in,
                              void* d_out, int out_size, void* d_ws, size_t ws_size,
                              hipStream_t stream) {
  const int*   item_seq     = (const int*)d_in[0];
  const int*   item_seq_len = (const int*)d_in[1];
  const float* emb          = (const float*)d_in[2];
  float* out = (float*)d_out;

  // workspace layout:
  //   emb_f16 : N_ITEMS*128 halves (25.6 MB)
  //   q_f16   : 4096*128 halves    ( 1.0 MB)
  //   pv      : N_SPLIT*4096*21 f32 (0.7 MB)
  //   pi      : N_SPLIT*4096*21 i32 (0.7 MB)
  _Float16* embh = (_Float16*)d_ws;
  _Float16* qh   = embh + (size_t)N_ITEMS * EMBED_DIM;
  float*    pv   = (float*)(qh + (size_t)BATCH * EMBED_DIM);
  int*      pi   = (int*)(pv + (size_t)N_SPLIT * BATCH * TOPK);

  cvt_f16_kernel<<<2048, 256, 0, stream>>>(emb, embh, N_ITEMS * EMBED_DIM);
  query_kernel<<<BATCH, EMBED_DIM, 0, stream>>>(item_seq, item_seq_len, emb, qh);
  score_topk_kernel<<<dim3(BATCH / BM, N_SPLIT), 256, 0, stream>>>(embh, qh, pv, pi);
  merge_kernel<<<(BATCH + 255) / 256, 256, 0, stream>>>(pv, pi, out);
}
